// GAT_35158602285297
// MI455X (gfx1250) — matrix operand order repaired and checked by reading, not compiled, not run on hardware
//
#include <hip/hip_runtime.h>
#include <hip/hip_bf16.h>
#include <math.h>

#define NNODES 50000
#define NEDGES 800000
#define ETOT   850000   // edges + self loops

typedef _Float16 v8h  __attribute__((ext_vector_type(8)));
typedef _Float16 v16h __attribute__((ext_vector_type(16)));
typedef float    v8f  __attribute__((ext_vector_type(8)));

// ---------------- helpers ----------------

__device__ __forceinline__ unsigned fkey(float f) {
  unsigned b = __float_as_uint(f);
  return b ^ ((b & 0x80000000u) ? 0xFFFFFFFFu : 0x80000000u);
}
__device__ __forceinline__ float funkey(unsigned k) {
  unsigned b = (k & 0x80000000u) ? (k ^ 0x80000000u) : ~k;
  return __uint_as_float(b);
}
__device__ __forceinline__ void edge_sd(const int* __restrict__ ei, int e, int& s, int& d) {
  if (e < NEDGES) { s = ei[e]; d = ei[NEDGES + e]; }
  else            { s = d = e - NEDGES; }      // self loops appended
}

// ---------------- f32 -> f16 convert ----------------

__global__ void k_cvt_f16(const float* __restrict__ src, _Float16* __restrict__ dst, int n) {
  int i = blockIdx.x * blockDim.x + threadIdx.x;
  if (i < n) dst[i] = (_Float16)src[i];
}

// ---------------- pack W [K,Ncols] into WMMA B lane order ----------------
// layout: [kt][nt][lane(32)][e(16)], B element (k = kt*32 + (lane>>4)*16 + e,
//          n = nt*16 + (lane&15))

__global__ void k_pack_w(const float* __restrict__ W, _Float16* __restrict__ Wp,
                         int K, int Ncols) {
  int idx = blockIdx.x * blockDim.x + threadIdx.x;
  if (idx >= K * Ncols) return;
  int e    = idx & 15;
  int lane = (idx >> 4) & 31;
  int t    = idx >> 9;          // kt*NT + nt
  int NT   = Ncols >> 4;
  int nt   = t % NT;
  int kt   = t / NT;
  int n    = nt * 16 + (lane & 15);
  int k    = kt * 32 + (lane >> 4) * 16 + e;
  Wp[idx]  = (_Float16)W[k * Ncols + n];
}

// ---------------- WMMA GEMM: C[M,Ncols] = A[M,K](f16) * Bpacked ----------------
// one wave per 16x16 C tile, 8 waves per block over consecutive M tiles

__global__ void __launch_bounds__(256)
k_gemm_wmma(const _Float16* __restrict__ A, const _Float16* __restrict__ Bp,
            float* __restrict__ C, int M, int K, int Ncols) {
  const int lane  = threadIdx.x & 31;
  const int wv    = threadIdx.x >> 5;
  const int tileM = blockIdx.x * 8 + wv;
  const int tileN = blockIdx.y;
  if (tileM * 16 >= M) return;                  // wave-uniform: EXEC stays full for WMMA
  const int mr  = lane & 15;
  const int khi = lane >> 4;
  const int NT  = Ncols >> 4;
  const _Float16* arow = A + (size_t)(tileM * 16 + mr) * K + khi * 16;
  const _Float16* bl   = Bp + ((size_t)tileN * 32 + lane) * 16;
  const size_t bstride = (size_t)NT * 512;      // halves per K-step block
  v8f acc = {};
  for (int kt = 0; kt < K; kt += 32) {
    v8h a0 = *(const v8h*)(arow + kt);          // K = khi*16 + 0..7
    v8h a1 = *(const v8h*)(arow + kt + 8);      // K = khi*16 + 8..15
    union { v16h v; v8h h[2]; } ua;
    ua.h[0] = a0; ua.h[1] = a1;
    v16h b = *(const v16h*)(bl + (size_t)(kt >> 5) * bstride);
    acc = __builtin_amdgcn_wmma_f32_16x16x32_f16(false, ua.v, false, b,
                                                 (short)0, acc, false, false);
  }
  float* cp = C + (size_t)(tileM * 16 + khi * 8) * Ncols + (size_t)tileN * 16 + mr;
#pragma unroll
  for (int i = 0; i < 8; i++) cp[(size_t)i * Ncols] = acc[i];
}

// ---------------- per-node attention scores: as[n,h] = sum_c h[n,h,c]*att[h,c] ----------------

__global__ void k_att(const float* __restrict__ h, const float* __restrict__ atts,
                      const float* __restrict__ attd, float* __restrict__ as,
                      float* __restrict__ ad, int H, int cshift) {
  int n = blockIdx.x * blockDim.x + threadIdx.x;
  if (n >= NNODES) return;
  int F = H << cshift;
  float s[4]  = {0.f, 0.f, 0.f, 0.f};
  float dd[4] = {0.f, 0.f, 0.f, 0.f};
  const float* row = h + (size_t)n * F;
  for (int f = 0; f < F; f++) {
    float v = row[f];
    int hh = f >> cshift;
    s[hh]  += v * atts[f];
    dd[hh] += v * attd[f];
  }
  for (int hh = 0; hh < H; hh++) { as[n * H + hh] = s[hh]; ad[n * H + hh] = dd[hh]; }
}

// ---------------- edge pass 1: leaky-relu alpha + segment max (monotonic-key atomicMax) ----------------

__global__ void k_edge_max(const int* __restrict__ ei, const float* __restrict__ as,
                           const float* __restrict__ ad, float* __restrict__ alpha,
                           unsigned* __restrict__ mkey, int H) {
  int e = blockIdx.x * blockDim.x + threadIdx.x;
  if (e >= ETOT) return;
  int s, d; edge_sd(ei, e, s, d);
  for (int hh = 0; hh < H; hh++) {
    float a = as[(size_t)s * H + hh] + ad[(size_t)d * H + hh];
    a = (a > 0.f) ? a : 0.2f * a;               // leaky relu, slope 0.2
    alpha[(size_t)e * H + hh] = a;
    atomicMax(&mkey[(size_t)d * H + hh], fkey(a));
  }
}

// ---------------- edge pass 2: ex = exp(alpha - m[dst]); segment sum ----------------

__global__ void k_edge_exp(const int* __restrict__ ei, float* __restrict__ alpha,
                           const unsigned* __restrict__ mkey, float* __restrict__ ssum,
                           int H) {
  int e = blockIdx.x * blockDim.x + threadIdx.x;
  if (e >= ETOT) return;
  int s, d; edge_sd(ei, e, s, d);
  (void)s;
  for (int hh = 0; hh < H; hh++) {
    float m  = funkey(mkey[(size_t)d * H + hh]);
    float ex = __expf(alpha[(size_t)e * H + hh] - m);
    alpha[(size_t)e * H + hh] = ex;             // overwrite alpha with exp value
    atomicAdd(&ssum[(size_t)d * H + hh], ex);
  }
}

// ---------------- edge pass 3: agg[dst] += h[src] * (ex / s[dst]); one wave per edge ----------------

__global__ void __launch_bounds__(256)
k_message(const int* __restrict__ ei, const float* __restrict__ hsrc,
          const float* __restrict__ ex, const float* __restrict__ ssum,
          float* __restrict__ agg, int H, int cshift) {
  int gid  = blockIdx.x * blockDim.x + threadIdx.x;
  int e    = gid >> 5;
  int lane = gid & 31;
  if (e >= ETOT) return;
  int s, d; edge_sd(ei, e, s, d);
  int F   = H << cshift;
  int per = F >> 5;                             // 8 (layer1) / 4 (layer2) channels per lane
  int c0  = lane * per;
  int hh  = c0 >> cshift;                       // per-lane channels never cross a head
  float coef = ex[(size_t)e * H + hh] / (ssum[(size_t)d * H + hh] + 1e-16f);
  const float* src = hsrc + (size_t)s * F + c0;
  float*       dst = agg  + (size_t)d * F + c0;
  for (int j = 0; j < per; j++) atomicAdd(&dst[j], src[j] * coef);
}

// ---------------- bias + ELU, emit f16 for next GEMM's A operand ----------------

__global__ void k_bias_elu_f16(const float* __restrict__ in, const float* __restrict__ b,
                               _Float16* __restrict__ out) {
  int i = blockIdx.x * blockDim.x + threadIdx.x;
  if (i >= NNODES * 256) return;
  float v = in[i] + b[i & 255];
  v = (v > 0.f) ? v : (__expf(v) - 1.f);
  out[i] = (_Float16)v;
}

// ---------------- bias + log_softmax over 128 channels, one wave per node ----------------

__global__ void __launch_bounds__(256)
k_logsoftmax(const float* __restrict__ agg, const float* __restrict__ b,
             float* __restrict__ out) {
  int gid  = blockIdx.x * blockDim.x + threadIdx.x;
  int n    = gid >> 5;
  int lane = gid & 31;
  if (n >= NNODES) return;
  const float* row = agg + (size_t)n * 128 + lane * 4;
  float v[4];
  float mx = -3.4e38f;
#pragma unroll
  for (int j = 0; j < 4; j++) { v[j] = row[j] + b[lane * 4 + j]; mx = fmaxf(mx, v[j]); }
  for (int o = 16; o > 0; o >>= 1) mx = fmaxf(mx, __shfl_xor(mx, o, 32));
  float se = 0.f;
#pragma unroll
  for (int j = 0; j < 4; j++) se += __expf(v[j] - mx);
  for (int o = 16; o > 0; o >>= 1) se += __shfl_xor(se, o, 32);
  float lse = mx + __logf(se);
  float* op = out + (size_t)n * 128 + lane * 4;
#pragma unroll
  for (int j = 0; j < 4; j++) op[j] = v[j] - lse;
}

// ---------------- launch ----------------

extern "C" void kernel_launch(void* const* d_in, const int* in_sizes, int n_in,
                              void* d_out, int out_size, void* d_ws, size_t ws_size,
                              hipStream_t stream) {
  (void)in_sizes; (void)n_in; (void)out_size; (void)ws_size;
  const float* x    = (const float*)d_in[0];
  const int*   ei   = (const int*)d_in[1];     // [2,E] (src row then dst row)
  const float* W1   = (const float*)d_in[2];
  const float* at1s = (const float*)d_in[3];
  const float* at1d = (const float*)d_in[4];
  const float* b1   = (const float*)d_in[5];
  const float* W2   = (const float*)d_in[6];
  const float* at2s = (const float*)d_in[7];
  const float* at2d = (const float*)d_in[8];
  const float* b2   = (const float*)d_in[9];
  float* out = (float*)d_out;

  char*  base = (char*)d_ws;
  size_t off  = 0;
  auto alloc = [&](size_t bytes) -> char* {
    char* r = base + off;
    off = (off + bytes + 255) & ~(size_t)255;
    return r;
  };
  _Float16* xh   = (_Float16*)alloc((size_t)NNODES * 256 * 2);
  _Float16* acth = (_Float16*)alloc((size_t)NNODES * 256 * 2);
  _Float16* W1p  = (_Float16*)alloc((size_t)256 * 256 * 2);
  _Float16* W2p  = (_Float16*)alloc((size_t)256 * 128 * 2);
  float*    h1   = (float*)alloc((size_t)NNODES * 256 * 4);
  float*    agg1 = (float*)alloc((size_t)NNODES * 256 * 4);
  float*    h2   = (float*)alloc((size_t)NNODES * 128 * 4);
  float*    agg2 = (float*)alloc((size_t)NNODES * 128 * 4);
  float*    sc1s = (float*)alloc((size_t)NNODES * 4 * 4);
  float*    sc1d = (float*)alloc((size_t)NNODES * 4 * 4);
  unsigned* mk1  = (unsigned*)alloc((size_t)NNODES * 4 * 4);
  float*    s1   = (float*)alloc((size_t)NNODES * 4 * 4);
  float*    sc2s = (float*)alloc((size_t)NNODES * 4);
  float*    sc2d = (float*)alloc((size_t)NNODES * 4);
  unsigned* mk2  = (unsigned*)alloc((size_t)NNODES * 4);
  float*    s2   = (float*)alloc((size_t)NNODES * 4);
  float*    ex1  = (float*)alloc((size_t)ETOT * 4 * 4);
  float*    ex2  = (float*)alloc((size_t)ETOT * 4);

  // re-init atomic targets every call (graph-capturable)
  hipMemsetAsync(agg1, 0, (size_t)NNODES * 256 * 4, stream);
  hipMemsetAsync(agg2, 0, (size_t)NNODES * 128 * 4, stream);
  hipMemsetAsync(mk1,  0, (size_t)NNODES * 4 * 4, stream);   // key 0 == -inf-most
  hipMemsetAsync(s1,   0, (size_t)NNODES * 4 * 4, stream);
  hipMemsetAsync(mk2,  0, (size_t)NNODES * 4, stream);
  hipMemsetAsync(s2,   0, (size_t)NNODES * 4, stream);

  const int T = 256;
  k_cvt_f16<<<(NNODES * 256 + T - 1) / T, T, 0, stream>>>(x, xh, NNODES * 256);
  k_pack_w<<<(256 * 256 + T - 1) / T, T, 0, stream>>>(W1, W1p, 256, 256);
  k_pack_w<<<(256 * 128 + T - 1) / T, T, 0, stream>>>(W2, W2p, 256, 128);

  const int mtiles = NNODES / 16;                 // 3125 exactly
  const int gx     = (mtiles + 7) / 8;            // 8 waves / block

  // ---- layer 1 ----
  dim3 g1(gx, 16);
  k_gemm_wmma<<<g1, 256, 0, stream>>>(xh, W1p, h1, NNODES, 256, 256);
  k_att<<<(NNODES + T - 1) / T, T, 0, stream>>>(h1, at1s, at1d, sc1s, sc1d, 4, 6);
  k_edge_max<<<(ETOT + T - 1) / T, T, 0, stream>>>(ei, sc1s, sc1d, ex1, mk1, 4);
  k_edge_exp<<<(ETOT + T - 1) / T, T, 0, stream>>>(ei, ex1, mk1, s1, 4);
  k_message<<<((size_t)ETOT * 32 + T - 1) / T, T, 0, stream>>>(ei, h1, ex1, s1, agg1, 4, 6);
  k_bias_elu_f16<<<(NNODES * 256 + T - 1) / T, T, 0, stream>>>(agg1, b1, acth);

  // ---- layer 2 ----
  dim3 g2(gx, 8);
  k_gemm_wmma<<<g2, 256, 0, stream>>>(acth, W2p, h2, NNODES, 256, 128);
  k_att<<<(NNODES + T - 1) / T, T, 0, stream>>>(h2, at2s, at2d, sc2s, sc2d, 1, 7);
  k_edge_max<<<(ETOT + T - 1) / T, T, 0, stream>>>(ei, sc2s, sc2d, ex2, mk2, 1);
  k_edge_exp<<<(ETOT + T - 1) / T, T, 0, stream>>>(ei, ex2, mk2, s2, 1);
  k_message<<<((size_t)ETOT * 32 + T - 1) / T, T, 0, stream>>>(ei, h2, ex2, s2, agg2, 1, 7);
  k_logsoftmax<<<((size_t)NNODES * 32 + T - 1) / T, T, 0, stream>>>(agg2, b2, out);
}
